// EnhancedPrototypeClusterer_45629732552857
// MI455X (gfx1250) — matrix operand order repaired and checked by reading, not compiled, not run on hardware
//
#include <hip/hip_runtime.h>
#include <hip/hip_bf16.h>

#define N_ROWS 32768
#define DIM    512
#define NCLS   1000
#define CPAD   1024

typedef __attribute__((ext_vector_type(16))) __bf16 v16bf;
typedef __attribute__((ext_vector_type(8)))  float  v8f;

union FragU { uint4 q[2]; v16bf v; };

__device__ __forceinline__ unsigned short f32_to_bf16(float f) {
    unsigned int u = __float_as_uint(f);
    unsigned int r = (u + 0x7FFFu + ((u >> 16) & 1u)) >> 16; // RNE
    return (unsigned short)r;
}

// ---------------- zero-init scratch region ----------------
__global__ void k_zero(unsigned int* __restrict__ p, long n) {
    long i = (long)blockIdx.x * blockDim.x + threadIdx.x;
    long stride = (long)gridDim.x * blockDim.x;
    for (; i < n; i += stride) p[i] = 0u;
}

// ------- normalize features -> bf16, scatter into per-class sums -------
__global__ void __launch_bounds__(256)
k_feat_norm_scatter(const float* __restrict__ feats,
                    const int* __restrict__ labels,
                    unsigned short* __restrict__ nf_bf16,
                    float* __restrict__ sums,
                    float* __restrict__ counts) {
    __shared__ float red[256];
    int row = blockIdx.x;
    int tid = threadIdx.x;
    const float* x = feats + (size_t)row * DIM;
    float v0 = x[tid], v1 = x[tid + 256];
    red[tid] = v0 * v0 + v1 * v1;
    __syncthreads();
    for (int s = 128; s > 0; s >>= 1) {
        if (tid < s) red[tid] += red[tid + s];
        __syncthreads();
    }
    float inv = 1.0f / fmaxf(sqrtf(red[0]), 1e-12f);
    float n0 = v0 * inv, n1 = v1 * inv;
    nf_bf16[(size_t)row * DIM + tid]       = f32_to_bf16(n0);
    nf_bf16[(size_t)row * DIM + tid + 256] = f32_to_bf16(n1);
    int lab = labels[row];
    atomicAdd(&sums[(size_t)lab * DIM + tid],       n0);
    atomicAdd(&sums[(size_t)lab * DIM + tid + 256], n1);
    if (tid == 0) atomicAdd(&counts[lab], 1.0f);
}

// ------- normalize prototypes -> bf16 into zero-padded [CPAD][DIM] -------
__global__ void __launch_bounds__(256)
k_proto_norm(const float* __restrict__ protos,
             unsigned short* __restrict__ npr_pad) {
    __shared__ float red[256];
    int row = blockIdx.x; // 0..NCLS-1
    int tid = threadIdx.x;
    const float* x = protos + (size_t)row * DIM;
    float v0 = x[tid], v1 = x[tid + 256];
    red[tid] = v0 * v0 + v1 * v1;
    __syncthreads();
    for (int s = 128; s > 0; s >>= 1) {
        if (tid < s) red[tid] += red[tid + s];
        __syncthreads();
    }
    float inv = 1.0f / fmaxf(sqrtf(red[0]), 1e-12f);
    npr_pad[(size_t)row * DIM + tid]       = f32_to_bf16(v0 * inv);
    npr_pad[(size_t)row * DIM + tid + 256] = f32_to_bf16(v1 * inv);
}

// ------- fused WMMA GEMM (16x1024 tile) + softmax + argmax -------
// Per K-step: issue A + all 8 B fragment loads as one clause, then 8 WMMAs.
// This lets later loads stay in flight while earlier WMMAs execute
// (partial s_wait_loadcnt), instead of load->wait0->wmma lockstep.
__global__ void __launch_bounds__(256)
k_sim_softmax(const unsigned short* __restrict__ nf,
              const unsigned short* __restrict__ npr,
              const float* __restrict__ temp_in,
              float* __restrict__ out_assign,
              float* __restrict__ out_probs) {
    __shared__ float sim[16][CPAD];            // 64 KB
    int tid  = threadIdx.x;
    int wave = tid >> 5, lane = tid & 31;
    int half = lane >> 4, l16 = lane & 15;
    int row0 = blockIdx.x * 16;

    const unsigned short* arow  = nf + (size_t)(row0 + l16) * DIM;
    int c0 = wave * 128;                       // 8 waves x 128 cols = 1024
    const unsigned short* bbase = npr + (size_t)(c0 + l16) * DIM;

    v8f acc[8];
#pragma unroll
    for (int t = 0; t < 8; ++t) acc[t] = (v8f){};

    for (int ks = 0; ks < 16; ++ks) {          // K = 512 in steps of 32
        int kb = ks * 32;
        FragU a, b[8];
        // --- batch-issue all loads for this K-step ---
        a.q[0] = *(const uint4*)(arow + kb + half * 8);       // k = kb+half*8 .. +7
        a.q[1] = *(const uint4*)(arow + kb + 16 + half * 8);  // k = kb+16+half*8 .. +7
#pragma unroll
        for (int t = 0; t < 8; ++t) {
            const unsigned short* brow = bbase + (size_t)t * 16 * DIM;
            b[t].q[0] = *(const uint4*)(brow + kb + half * 8);
            b[t].q[1] = *(const uint4*)(brow + kb + 16 + half * 8);
        }
        // --- then compute ---
#pragma unroll
        for (int t = 0; t < 8; ++t)
            acc[t] = __builtin_amdgcn_wmma_f32_16x16x32_bf16(
                false, a.v, false, b[t].v, (short)0, acc[t], false, false);
    }

#pragma unroll
    for (int t = 0; t < 8; ++t)
#pragma unroll
        for (int v = 0; v < 8; ++v)
            sim[v + 8 * half][c0 + t * 16 + l16] = acc[t][v];
    __syncthreads();

    // softmax over c<1000 : 16 threads per row, 64 cols each
    float traw = temp_in[0];
    float temp = (traw > 20.0f ? traw : log1pf(__expf(traw))) + 0.01f;
    int r    = tid >> 4;         // row 0..15
    int t16  = tid & 15;
    int cbeg = t16 * 64;
    int cend = min(cbeg + 64, NCLS);

    float bmax = -3.4e38f; int bidx = 0;
    for (int c = cbeg; c < cend; ++c) {
        float v = sim[r][c];
        if (v > bmax) { bmax = v; bidx = c; }
    }
#pragma unroll
    for (int m = 8; m >= 1; m >>= 1) {
        float om = __shfl_xor(bmax, m, 16);
        int   oi = __shfl_xor(bidx, m, 16);
        if (om > bmax || (om == bmax && oi < bidx)) { bmax = om; bidx = oi; }
    }
    float inv_t = 1.0f / temp;
    float lsum = 0.0f;
    for (int c = cbeg; c < cend; ++c) {
        float e = __expf((sim[r][c] - bmax) * inv_t);
        sim[r][c] = e;
        lsum += e;
    }
#pragma unroll
    for (int m = 8; m >= 1; m >>= 1) lsum += __shfl_xor(lsum, m, 16);
    float invs = 1.0f / lsum;
    float* prow = out_probs + (size_t)(row0 + r) * NCLS;
    for (int c = cbeg; c < cend; ++c) prow[c] = sim[r][c] * invs;
    if (t16 == 0) out_assign[row0 + r] = (float)bidx;
}

// ------- prototype regularization: WMMA npr.nprT, reduce to 2 scalars -------
__global__ void __launch_bounds__(256)
k_proto_reg(const unsigned short* __restrict__ npr,
            float* __restrict__ accum /* [0]=sum sqdiff, [1]=sum */) {
    __shared__ float sA[256], sB[256];
    int tid = threadIdx.x, wave = tid >> 5, lane = tid & 31;
    int half = lane >> 4, l16 = lane & 15;
    int tile = blockIdx.x * 8 + wave;          // 0..4095 over 64x64 tiles
    int tm = tile >> 6, tn = tile & 63;
    const unsigned short* arow = npr + (size_t)(tm * 16 + l16) * DIM;
    const unsigned short* brow = npr + (size_t)(tn * 16 + l16) * DIM;

    v8f acc = (v8f){};
#pragma unroll 4
    for (int ks = 0; ks < 16; ++ks) {
        int kb = ks * 32;
        FragU a, b;        // fresh locals per unrolled copy -> renamed regs
        a.q[0] = *(const uint4*)(arow + kb + half * 8);
        a.q[1] = *(const uint4*)(arow + kb + 16 + half * 8);
        b.q[0] = *(const uint4*)(brow + kb + half * 8);
        b.q[1] = *(const uint4*)(brow + kb + 16 + half * 8);
        acc = __builtin_amdgcn_wmma_f32_16x16x32_bf16(
            false, a.v, false, b.v, (short)0, acc, false, false);
    }
    float ssd = 0.0f, sv = 0.0f;
    int n = tn * 16 + l16;
#pragma unroll
    for (int v = 0; v < 8; ++v) {
        int m = tm * 16 + v + 8 * half;
        if (m < NCLS && n < NCLS) {
            float val = acc[v];
            float d = val - ((m == n) ? 1.0f : 0.0f);
            ssd += d * d;
            sv  += val;
        }
    }
    sA[tid] = ssd; sB[tid] = sv;
    __syncthreads();
    for (int s = 128; s > 0; s >>= 1) {
        if (tid < s) { sA[tid] += sA[tid + s]; sB[tid] += sB[tid + s]; }
        __syncthreads();
    }
    if (tid == 0) { atomicAdd(&accum[0], sA[0]); atomicAdd(&accum[1], sB[0]); }
}

// ------- EMA update + reg-loss scalar -------
__global__ void __launch_bounds__(256)
k_ema_finalize(const float* __restrict__ ema_old,
               const float* __restrict__ sums,
               const float* __restrict__ counts,
               const float* __restrict__ accum,
               float* __restrict__ out_ema,
               float* __restrict__ out_reg) {
    int idx = blockIdx.x * blockDim.x + threadIdx.x;
    if (idx == 0) {
        float inv = 1.0f / ((float)NCLS * (float)NCLS);
        float div_loss = accum[0] * inv;
        float uni = accum[1] * inv - 0.5f;
        out_reg[0] = 0.1f * (div_loss + uni * uni);
    }
    if (idx < NCLS * DIM) {
        int c = idx / DIM;
        float cnt  = counts[c];
        float e    = ema_old[idx];
        float mean = sums[idx] / fmaxf(cnt, 1.0f);
        out_ema[idx] = (cnt > 0.0f) ? 0.99f * e + 0.01f * mean : e;
    }
}

extern "C" void kernel_launch(void* const* d_in, const int* in_sizes, int n_in,
                              void* d_out, int out_size, void* d_ws, size_t ws_size,
                              hipStream_t stream) {
    const float* features    = (const float*)d_in[0];
    const int*   labels      = (const int*)d_in[1];
    const float* prototypes  = (const float*)d_in[2];
    const float* temperature = (const float*)d_in[3];
    const float* ema_old     = (const float*)d_in[4];

    char* ws = (char*)d_ws;
    unsigned short* nf_bf16 = (unsigned short*)ws;                    // N*D bf16
    size_t off = (size_t)N_ROWS * DIM * sizeof(unsigned short);
    unsigned short* npr_pad = (unsigned short*)(ws + off);            // CPAD*D bf16 (zero pad)
    off += (size_t)CPAD * DIM * sizeof(unsigned short);
    float* sums = (float*)(ws + off);                                 // NCLS*D f32
    off += (size_t)NCLS * DIM * sizeof(float);
    float* counts = (float*)(ws + off);                               // NCLS f32
    off += (size_t)NCLS * sizeof(float);
    float* accum = (float*)(ws + off);                                // 2 f32

    float* out_assign = (float*)d_out;                                // N
    float* out_probs  = out_assign + N_ROWS;                          // N*NCLS
    float* out_reg    = out_probs + (size_t)N_ROWS * NCLS;            // 1
    float* out_ema    = out_reg + 1;                                  // NCLS*DIM

    // zero npr_pad + sums + counts + accum (contiguous region)
    size_t zero_words = ((size_t)CPAD * DIM * 2 + (size_t)NCLS * DIM * 4 +
                         (size_t)NCLS * 4 + 8) / 4;
    k_zero<<<1024, 256, 0, stream>>>((unsigned int*)npr_pad, (long)zero_words);

    k_feat_norm_scatter<<<N_ROWS, 256, 0, stream>>>(features, labels, nf_bf16, sums, counts);
    k_proto_norm<<<NCLS, 256, 0, stream>>>(prototypes, npr_pad);
    k_sim_softmax<<<N_ROWS / 16, 256, 0, stream>>>(nf_bf16, npr_pad, temperature,
                                                   out_assign, out_probs);
    k_proto_reg<<<512, 256, 0, stream>>>(npr_pad, accum);
    k_ema_finalize<<<(NCLS * DIM + 255) / 256, 256, 0, stream>>>(
        ema_old, sums, counts, accum, out_ema, out_reg);
}
